// MultiHeadAttention_44530220925320
// MI455X (gfx1250) — compile-verified, matrix operand order checked by reading, not run on hardware
//
#include <hip/hip_runtime.h>
#include <hip/hip_bf16.h>

// ---------------------------------------------------------------------------
// MHA for MI455X (gfx1250): bf16 WMMA (16x16x32, f32 accum) everywhere.
//   q/k/v:[1,4096,768] f32, W*:[768,768] f32, b*:[768] f32 -> out [1,4096,768] f32
// Pipeline: cvt(f32->bf16) -> 3x GEMM proj -> flash attention -> GEMM out-proj.
// GEMM is register double-buffered so WMMAs overlap fragment loads instead of
// hitting s_wait_loadcnt 0 per instruction; global_prefetch_b8 warms L2 ahead.
// Workspace: needs ~49 MB of d_ws.
// ---------------------------------------------------------------------------

#define SEQLEN 4096
#define DMODEL 768
#define NHEAD  12
#define HDIM   64

typedef __bf16 v16bf __attribute__((ext_vector_type(16)));
typedef __bf16 v8bf  __attribute__((ext_vector_type(8)));
typedef float  v8f   __attribute__((ext_vector_type(8)));

__device__ __forceinline__ __bf16 f32_to_bf16(float f) {
  unsigned u = __builtin_bit_cast(unsigned, f);
  u += 0x7FFFu + ((u >> 16) & 1u);            // round-to-nearest-even
  unsigned short h = (unsigned short)(u >> 16);
  return __builtin_bit_cast(__bf16, h);
}

__device__ __forceinline__ v8f wmma_bf16(v16bf a, v16bf b, v8f c) {
  // D = A(16x32 bf16) x B(32x16 bf16) + C(16x16 f32)  -> v_wmma_f32_16x16x32_bf16
  return __builtin_amdgcn_wmma_f32_16x16x32_bf16(
      /*neg_a=*/false, a, /*neg_b=*/false, b,
      /*c_mod=*/(short)0, c, /*reuse_a=*/false, /*reuse_b=*/false);
}

// Load a 16x32 bf16 operand fragment (A layout; B uses the identical pattern
// because the B-operand lane n holds "row n" of the K-contiguous source).
// ISA 7.12.2: lanes 0-15 -> rows 0-15, K 0-7 & 16-23; lanes 16-31 -> K 8-15 & 24-31.
__device__ __forceinline__ v16bf load_frag16x32(const __bf16* base, int ld) {
  int lane = threadIdx.x & 31;
  int row  = lane & 15;
  int kb   = (lane >> 4) << 3;                 // 0 or 8
  const __bf16* p = base + row * ld + kb;
  v8bf lo = *(const v8bf*)(p);                 // K = kb .. kb+7      (16B load)
  v8bf hi = *(const v8bf*)(p + 16);            // K = kb+16 .. kb+23  (16B load)
  v16bf r;
#pragma unroll
  for (int j = 0; j < 8; ++j) { r[j] = lo[j]; r[8 + j] = hi[j]; }
  return r;
}

// Row reductions across the 16-lane C-layout groups (wave32: xor<16 stays in-group).
__device__ __forceinline__ float rowred_max(float x) {
#pragma unroll
  for (int s = 1; s < 16; s <<= 1) x = fmaxf(x, __shfl_xor(x, s, 32));
  return x;
}
__device__ __forceinline__ float rowred_sum(float x) {
#pragma unroll
  for (int s = 1; s < 16; s <<= 1) x += __shfl_xor(x, s, 32);
  return x;
}

// ---------------------------------------------------------------------------
__global__ void cvt_f32_bf16(const float* __restrict__ in, __bf16* __restrict__ out, int n) {
  int i = blockIdx.x * blockDim.x + threadIdx.x;
  if (i < n) out[i] = f32_to_bf16(in[i]);
}

// ---------------------------------------------------------------------------
// C[M,N] = A[M,K] @ W[N,K]^T + bias.
// Block = 256 thr = 8 waves (4 M x 2 N); wave tile = 32x64 (8 WMMA accums),
// block tile = 128x128. K loop is register double-buffered: next iteration's
// 6 fragments (12x b128) issue before this iteration's 8 WMMAs so the matrix
// pipe overlaps memory latency. Grid = (N/128, M/128).
__global__ __launch_bounds__(256) void gemm_xwt(
    const __bf16* __restrict__ A, const __bf16* __restrict__ W,
    const float* __restrict__ bias, float* __restrict__ outf,
    __bf16* __restrict__ outb, int M, int N, int K)
{
  int w    = threadIdx.x >> 5;
  int lane = threadIdx.x & 31;
  int m0 = blockIdx.y * 128 + (w & 3) * 32;
  int n0 = blockIdx.x * 128 + (w >> 2) * 64;

  const __bf16* Arow0 = A + (size_t)m0 * K;
  const __bf16* Arow1 = A + (size_t)(m0 + 16) * K;
  const __bf16* Wrow[4];
#pragma unroll
  for (int t = 0; t < 4; ++t) Wrow[t] = W + (size_t)(n0 + 16 * t) * K;

  v8f acc[2][4] = {};

  // prologue: fragments for kt = 0
  v16bf a0 = load_frag16x32(Arow0, K);
  v16bf a1 = load_frag16x32(Arow1, K);
  v16bf b0 = load_frag16x32(Wrow[0], K);
  v16bf b1 = load_frag16x32(Wrow[1], K);
  v16bf b2 = load_frag16x32(Wrow[2], K);
  v16bf b3 = load_frag16x32(Wrow[3], K);

  for (int kt = 32; kt < K; kt += 32) {
    // warm L2 two tiles ahead (global_prefetch_b8)
    if (kt + 64 < K) {
      __builtin_prefetch(Arow0 + kt + 64 + (lane & 15) * K, 0, 0);
      __builtin_prefetch(Wrow[0] + kt + 64 + (lane & 15) * K, 0, 0);
    }
    // issue next iteration's fragment loads first
    v16bf na0 = load_frag16x32(Arow0 + kt, K);
    v16bf na1 = load_frag16x32(Arow1 + kt, K);
    v16bf nb0 = load_frag16x32(Wrow[0] + kt, K);
    v16bf nb1 = load_frag16x32(Wrow[1] + kt, K);
    v16bf nb2 = load_frag16x32(Wrow[2] + kt, K);
    v16bf nb3 = load_frag16x32(Wrow[3] + kt, K);

    // compute on current fragments (B reused across both A tiles)
    acc[0][0] = wmma_bf16(a0, b0, acc[0][0]);
    acc[1][0] = wmma_bf16(a1, b0, acc[1][0]);
    acc[0][1] = wmma_bf16(a0, b1, acc[0][1]);
    acc[1][1] = wmma_bf16(a1, b1, acc[1][1]);
    acc[0][2] = wmma_bf16(a0, b2, acc[0][2]);
    acc[1][2] = wmma_bf16(a1, b2, acc[1][2]);
    acc[0][3] = wmma_bf16(a0, b3, acc[0][3]);
    acc[1][3] = wmma_bf16(a1, b3, acc[1][3]);

    a0 = na0; a1 = na1; b0 = nb0; b1 = nb1; b2 = nb2; b3 = nb3;
  }
  // epilogue WMMAs for the last K tile
  acc[0][0] = wmma_bf16(a0, b0, acc[0][0]);
  acc[1][0] = wmma_bf16(a1, b0, acc[1][0]);
  acc[0][1] = wmma_bf16(a0, b1, acc[0][1]);
  acc[1][1] = wmma_bf16(a1, b1, acc[1][1]);
  acc[0][2] = wmma_bf16(a0, b2, acc[0][2]);
  acc[1][2] = wmma_bf16(a1, b2, acc[1][2]);
  acc[0][3] = wmma_bf16(a0, b3, acc[0][3]);
  acc[1][3] = wmma_bf16(a1, b3, acc[1][3]);

  int colg   = lane & 15;
  int rowoff = (lane >> 4) * 8;
#pragma unroll
  for (int i = 0; i < 2; ++i) {
#pragma unroll
    for (int t = 0; t < 4; ++t) {
      int n = n0 + 16 * t + colg;
      float bv = bias[n];
#pragma unroll
      for (int r = 0; r < 8; ++r) {
        int m = m0 + 16 * i + rowoff + r;
        float v = acc[i][t][r] + bv;
        if (outf) outf[(size_t)m * N + n] = v;
        if (outb) outb[(size_t)m * N + n] = f32_to_bf16(v);
      }
    }
  }
}

// ---------------------------------------------------------------------------
// Flash attention, one head-slice per block: grid = (SEQ/64 row tiles, NHEAD),
// block = 128 thr = 4 waves; wave w owns 16 query rows. Online softmax in f32.
__global__ __launch_bounds__(128) void flash_attn(
    const __bf16* __restrict__ Q, const __bf16* __restrict__ K,
    const __bf16* __restrict__ V, __bf16* __restrict__ O)
{
  __shared__ __bf16 ldsVt[HDIM * 64];        // 8 KB  V^T tile: [hd][key]
  __shared__ __bf16 ldsP[4][16 * 64];        // 8 KB  per-wave P tile (C->A relayout)

  int tid  = threadIdx.x;
  int w    = tid >> 5;
  int lane = tid & 31;
  int h    = blockIdx.y;
  int qrow0 = blockIdx.x * 64 + w * 16;

  const __bf16* qbase = Q + (size_t)qrow0 * DMODEL + h * HDIM;
  v16bf qf0 = load_frag16x32(qbase,      DMODEL);   // Hd 0..31
  v16bf qf1 = load_frag16x32(qbase + 32, DMODEL);   // Hd 32..63

  v8f acc[4] = {};
  float mrow[8], lrow[8];
#pragma unroll
  for (int r = 0; r < 8; ++r) { mrow[r] = -1e30f; lrow[r] = 0.f; }

  const float scale = 0.125f;                 // 1/sqrt(64)
  int colg   = lane & 15;
  int rowoff = (lane >> 4) * 8;

  for (int kb = 0; kb < SEQLEN; kb += 64) {
    // warm L2 for the next key block's K and V rows (global_prefetch_b8)
    if (kb + 64 < SEQLEN) {
      const __bf16* pk = K + (size_t)(kb + 64 + (tid & 63)) * DMODEL + h * HDIM;
      const __bf16* pv = V + (size_t)(kb + 64 + (tid & 63)) * DMODEL + h * HDIM;
      __builtin_prefetch(pk, 0, 0);
      __builtin_prefetch(pv, 0, 0);
    }

    // --- cooperative transposed stage of V tile [64 keys x 64 hd] -> Vt[hd][key]
    for (int c = tid; c < 512; c += 128) {    // 512 chunks of 8 halves (16B loads)
      int key = c >> 3;
      int hd0 = (c & 7) << 3;
      v8bf vv = *(const v8bf*)(V + (size_t)(kb + key) * DMODEL + h * HDIM + hd0);
#pragma unroll
      for (int j = 0; j < 8; ++j) ldsVt[(hd0 + j) * 64 + key] = vv[j];
    }
    __syncthreads();

    // --- scores S[16 x 64] = Q_tile @ K_tile^T (K rows are Hd-contiguous)
    v8f s[4];
#pragma unroll
    for (int t = 0; t < 4; ++t) {
      const __bf16* kbase = K + (size_t)(kb + 16 * t) * DMODEL + h * HDIM;
      v16bf b0 = load_frag16x32(kbase,      DMODEL);
      v16bf b1 = load_frag16x32(kbase + 32, DMODEL);
      v8f z = {};
      z    = wmma_bf16(qf0, b0, z);
      s[t] = wmma_bf16(qf1, b1, z);
    }

    // --- online softmax (row stats over 16 lanes x 4 tiles)
    float newm[8];
#pragma unroll
    for (int r = 0; r < 8; ++r) {
      float mx = fmaxf(fmaxf(s[0][r], s[1][r]), fmaxf(s[2][r], s[3][r]));
      mx = rowred_max(mx) * scale;
      newm[r] = fmaxf(mrow[r], mx);
    }
    float rsum[8];
#pragma unroll
    for (int r = 0; r < 8; ++r) rsum[r] = 0.f;
#pragma unroll
    for (int t = 0; t < 4; ++t) {
#pragma unroll
      for (int r = 0; r < 8; ++r) {
        float p = __expf(s[t][r] * scale - newm[r]);
        rsum[r] += p;
        ldsP[w][(rowoff + r) * 64 + t * 16 + colg] = f32_to_bf16(p);
      }
    }
#pragma unroll
    for (int r = 0; r < 8; ++r) {
      float bs = rowred_sum(rsum[r]);
      float alpha = __expf(mrow[r] - newm[r]);
      lrow[r] = lrow[r] * alpha + bs;
      mrow[r] = newm[r];
#pragma unroll
      for (int t = 0; t < 4; ++t) acc[t][r] *= alpha;
    }

    // --- P(16x64 bf16) @ V(64x64): A-frags from ldsP, B-frags from ldsVt rows
    v16bf p0 = load_frag16x32(&ldsP[w][0],  64);    // keys 0..31
    v16bf p1 = load_frag16x32(&ldsP[w][32], 64);    // keys 32..63
#pragma unroll
    for (int t = 0; t < 4; ++t) {
      v16bf b0 = load_frag16x32(&ldsVt[(t * 16) * 64 + 0],  64);
      v16bf b1 = load_frag16x32(&ldsVt[(t * 16) * 64 + 32], 64);
      acc[t] = wmma_bf16(p0, b0, acc[t]);
      acc[t] = wmma_bf16(p1, b1, acc[t]);
    }
    __syncthreads();
  }

  // --- finalize: O = acc / l   (concat-head layout [SEQ, DMODEL] in bf16)
#pragma unroll
  for (int t = 0; t < 4; ++t)
#pragma unroll
    for (int r = 0; r < 8; ++r) {
      float v = acc[t][r] / lrow[r];
      O[(size_t)(qrow0 + rowoff + r) * DMODEL + h * HDIM + t * 16 + colg] =
          f32_to_bf16(v);
    }
}

// ---------------------------------------------------------------------------
extern "C" void kernel_launch(void* const* d_in, const int* in_sizes, int n_in,
                              void* d_out, int out_size, void* d_ws, size_t ws_size,
                              hipStream_t stream) {
  (void)in_sizes; (void)n_in; (void)out_size; (void)ws_size;
  const float* q  = (const float*)d_in[0];
  const float* k  = (const float*)d_in[1];
  const float* v  = (const float*)d_in[2];
  const float* Wq = (const float*)d_in[3];
  const float* bq = (const float*)d_in[4];
  const float* Wk = (const float*)d_in[5];
  const float* bk = (const float*)d_in[6];
  const float* Wv = (const float*)d_in[7];
  const float* bv = (const float*)d_in[8];
  const float* Wo = (const float*)d_in[9];
  const float* bo = (const float*)d_in[10];

  // workspace carve-up (all 16B-aligned); total = 7*SB + 4*WB ~= 48.8 MB
  char* ws = (char*)d_ws;
  const size_t SB = (size_t)SEQLEN * DMODEL * 2;   // 6,291,456 B
  const size_t WB = (size_t)DMODEL * DMODEL * 2;   // 1,179,648 B
  __bf16* qb  = (__bf16*)(ws + 0 * SB);
  __bf16* kbm = (__bf16*)(ws + 1 * SB);
  __bf16* vbm = (__bf16*)(ws + 2 * SB);
  __bf16* Qp  = (__bf16*)(ws + 3 * SB);
  __bf16* Kp  = (__bf16*)(ws + 4 * SB);
  __bf16* Vp  = (__bf16*)(ws + 5 * SB);
  __bf16* Ab  = (__bf16*)(ws + 6 * SB);
  __bf16* wqb = (__bf16*)(ws + 7 * SB + 0 * WB);
  __bf16* wkb = (__bf16*)(ws + 7 * SB + 1 * WB);
  __bf16* wvb = (__bf16*)(ws + 7 * SB + 2 * WB);
  __bf16* wob = (__bf16*)(ws + 7 * SB + 3 * WB);

  const int nS = SEQLEN * DMODEL, nW = DMODEL * DMODEL;
  // fp32 -> bf16 conversions
  cvt_f32_bf16<<<(nS + 255) / 256, 256, 0, stream>>>(q,  qb,  nS);
  cvt_f32_bf16<<<(nS + 255) / 256, 256, 0, stream>>>(k,  kbm, nS);
  cvt_f32_bf16<<<(nS + 255) / 256, 256, 0, stream>>>(v,  vbm, nS);
  cvt_f32_bf16<<<(nW + 255) / 256, 256, 0, stream>>>(Wq, wqb, nW);
  cvt_f32_bf16<<<(nW + 255) / 256, 256, 0, stream>>>(Wk, wkb, nW);
  cvt_f32_bf16<<<(nW + 255) / 256, 256, 0, stream>>>(Wv, wvb, nW);
  cvt_f32_bf16<<<(nW + 255) / 256, 256, 0, stream>>>(Wo, wob, nW);

  // QKV projections (bf16 outputs); grid = (N/128, M/128)
  dim3 gg(DMODEL / 128, SEQLEN / 128), gb(256);
  gemm_xwt<<<gg, gb, 0, stream>>>(qb,  wqb, bq, nullptr, Qp, SEQLEN, DMODEL, DMODEL);
  gemm_xwt<<<gg, gb, 0, stream>>>(kbm, wkb, bk, nullptr, Kp, SEQLEN, DMODEL, DMODEL);
  gemm_xwt<<<gg, gb, 0, stream>>>(vbm, wvb, bv, nullptr, Vp, SEQLEN, DMODEL, DMODEL);

  // fused softmax(QK^T/8)V per head, concat-head output in bf16
  flash_attn<<<dim3(SEQLEN / 64, NHEAD), 128, 0, stream>>>(Qp, Kp, Vp, Ab);

  // output projection -> fp32 d_out
  gemm_xwt<<<gg, gb, 0, stream>>>(Ab, wob, bo, (float*)d_out, nullptr,
                                  SEQLEN, DMODEL, DMODEL);
}
